// ALiBiSelfAttention_21534966022953
// MI455X (gfx1250) — matrix operand order repaired and checked by reading, not compiled, not run on hardware
//
#include <hip/hip_runtime.h>
#include <hip/hip_bf16.h>

// ALiBi causal self-attention for MI455X (gfx1250, wave32, WMMA).
// B=2, T=2048, D=1024, H=16, hd=64. All matmuls via v_wmma_f32_16x16x32_bf16.
//
// d_out layout (tuple return): [ out: B*T*D fp32 | attn: B*H*T*T fp32 ]
// Workspace layout (bytes), total 48 MiB:
//   xb   @ 0         : B*T*D bf16           (8 MiB)
//   wqb  @ 8 MiB     : D*D bf16             (2 MiB)
//   wkb  @ 10 MiB    : D*D bf16
//   wvb  @ 12 MiB    : D*D bf16
//   wob  @ 14 MiB    : D*D bf16
//   qb   @ 16 MiB    : [B*H, T, 64] bf16    (8 MiB)
//   kb   @ 24 MiB    : [B*H, T, 64] bf16
//   vb   @ 32 MiB    : [B*H, T, 64] bf16
//   ctx  @ 40 MiB    : [B*T, D] bf16        (8 MiB)

#define BATCH 2
#define SEQ   2048
#define DIM   1024
#define HEADS 16
#define HD    64
#define MROWS (BATCH * SEQ)   // 4096
#define BH    (BATCH * HEADS) // 32
#define ALIBI 1.0f

typedef __attribute__((ext_vector_type(16))) __bf16 v16bf;
typedef __attribute__((ext_vector_type(8)))  float  v8f;
typedef unsigned short u16;
typedef unsigned int   u32;

union Frag16 {           // 16 bf16 = 32 bytes = 8 VGPRs
    v16bf v;
    u16   u[16];
    uint4 q[2];
};

static __device__ __forceinline__ u16 f32_to_bf16(float f) {
    u32 x = __float_as_uint(f);
    u32 lsb = (x >> 16) & 1u;
    x += 0x7fffu + lsb;                       // round-to-nearest-even
    return (u16)(x >> 16);
}

// ---------------------------------------------------------------- convert --
__global__ void cvt_f32_to_bf16(const float* __restrict__ in,
                                u16* __restrict__ out, int n4) {
    int i = blockIdx.x * blockDim.x + threadIdx.x;
    if (i >= n4) return;
    float4 f = ((const float4*)in)[i];
    u32 lo = (u32)f32_to_bf16(f.x) | ((u32)f32_to_bf16(f.y) << 16);
    u32 hi = (u32)f32_to_bf16(f.z) | ((u32)f32_to_bf16(f.w) << 16);
    ((uint2*)out)[i] = make_uint2(lo, hi);
}

// ------------------------------------------------------------- projection --
// C[m,n] = sum_k Xb[m,k] * Wb[n,k] + bias[n]   (i.e. y = x @ W.T + b)
// Xb: [MROWS, DIM] bf16 row-major. Wb: [DIM, DIM] bf16 row-major.
// mode 0: write bf16 head-major [B*H, T, 64] (for Q/K/V)
// mode 1: write fp32 row-major [MROWS, DIM]  (final projection)
// grid (MROWS/16, DIM/256), block 128: 4 waves, each wave owns a 16x64
// output strip (4 register-blocked 16x16 accumulators sharing one A frag).
__global__ void proj_wmma(const u16* __restrict__ Xb, const u16* __restrict__ Wb,
                          const float* __restrict__ bias,
                          u16* __restrict__ outb, float* __restrict__ outf,
                          int mode) {
    __shared__ u16 As[16 * 64];                 // 16 rows x 64-K chunk, 2 KiB
    const int m0   = blockIdx.x * 16;
    const int wave = threadIdx.x >> 5;
    const int lane = threadIdx.x & 31;
    const int half = lane >> 4;
    const int l    = lane & 15;
    const int n0   = blockIdx.y * 256 + wave * 64;

    v8f acc[4] = {{}, {}, {}, {}};
    for (int ks = 0; ks < DIM; ks += 64) {
        // stage A tile: 1024 bf16 by 128 threads, one b128 each
        {
            int e = threadIdx.x * 8;
            int r = e >> 6, c = e & 63;
            *(uint4*)&As[e] = *(const uint4*)&Xb[(m0 + r) * DIM + ks + c];
        }
        // pull the next K-chunk of the weight rows toward the WGP caches
        __builtin_prefetch(&Wb[(n0 + l) * DIM + ks + 64], 0, 3);
        __builtin_prefetch(&Wb[(n0 + 32 + l) * DIM + ks + 64], 0, 3);
        __syncthreads();
        for (int kk = 0; kk < 64; kk += 32) {
            Frag16 a;
            // A fragment (16x32 bf16): lane l row M=l; half 0 -> K 0..15, half 1 -> K 16..31
            a.q[0] = *(const uint4*)&As[l * 64 + kk + half * 16];
            a.q[1] = *(const uint4*)&As[l * 64 + kk + half * 16 + 8];
#pragma unroll
            for (int s = 0; s < 4; ++s) {
                Frag16 b;
                // B fragment (32x16): lane l col N; half selects K 0..15/16..31
                const u16* wr = &Wb[(n0 + s * 16 + l) * DIM + ks + kk + half * 16];
                b.q[0] = *(const uint4*)&wr[0];
                b.q[1] = *(const uint4*)&wr[8];
                acc[s] = __builtin_amdgcn_wmma_f32_16x16x32_bf16(
                             false, a.v, false, b.v, (short)0, acc[s], false, false);
            }
        }
        __syncthreads();
    }
    // epilogue: element (M = r + 8*half, N = l)
#pragma unroll
    for (int s = 0; s < 4; ++s) {
        const int n = n0 + s * 16 + l;
        const float bv = bias[n];
        for (int r = 0; r < 8; ++r) {
            int m = m0 + r + half * 8;
            float val = acc[s][r] + bv;
            if (mode == 0) {
                int h = n >> 6, d = n & 63;
                int bi = m >> 11, t = m & (SEQ - 1);
                outb[((bi * HEADS + h) * SEQ + t) * HD + d] = f32_to_bf16(val);
            } else {
                outf[m * DIM + n] = val;
            }
        }
    }
}

// ----------------------------------------------------------------- scores --
// raw scores(i,j) = (q_i . k_j)/8 - ALIBI*(i-j) for j<=i; -inf inside the
// diagonal tile for j>i (softmax pass overwrites masked region with 0).
// grid (SEQ/16, BH), block 128; waves split the key tiles, A frags hoisted.
__global__ void scores_wmma(const u16* __restrict__ qb, const u16* __restrict__ kb,
                            float* __restrict__ attn) {
    const int q0   = blockIdx.x * 16;
    const int bh   = blockIdx.y;
    const int wave = threadIdx.x >> 5;
    const int lane = threadIdx.x & 31;
    const int half = lane >> 4;
    const int l    = lane & 15;

    // A fragments (query tile, K = hd = 64 -> two 16x32 steps), hoisted
    Frag16 a0, a1;
    const u16* qrow = &qb[(bh * SEQ + q0 + l) * HD];
    a0.q[0] = *(const uint4*)&qrow[half * 16];
    a0.q[1] = *(const uint4*)&qrow[half * 16 + 8];
    a1.q[0] = *(const uint4*)&qrow[32 + half * 16];
    a1.q[1] = *(const uint4*)&qrow[32 + half * 16 + 8];

    for (int kt0 = wave * 16; kt0 < q0 + 16; kt0 += 64) {
        Frag16 b0, b1;
        const u16* krow = &kb[(bh * SEQ + kt0 + l) * HD];
        b0.q[0] = *(const uint4*)&krow[half * 16];
        b0.q[1] = *(const uint4*)&krow[half * 16 + 8];
        b1.q[0] = *(const uint4*)&krow[32 + half * 16];
        b1.q[1] = *(const uint4*)&krow[32 + half * 16 + 8];
        v8f acc = {};
        acc = __builtin_amdgcn_wmma_f32_16x16x32_bf16(
                  false, a0.v, false, b0.v, (short)0, acc, false, false);
        acc = __builtin_amdgcn_wmma_f32_16x16x32_bf16(
                  false, a1.v, false, b1.v, (short)0, acc, false, false);
        for (int r = 0; r < 8; ++r) {
            int i = q0 + r + half * 8;
            int j = kt0 + l;
            float s = (j <= i) ? (acc[r] * 0.125f - ALIBI * (float)(i - j))
                               : -__builtin_inff();
            attn[((size_t)bh * SEQ + i) * SEQ + j] = s;
        }
    }
}

// ---------------------------------------------------------------- softmax --
// one wave per row; writes the whole row (exact zeros where j > i).
__global__ void softmax_rows(float* __restrict__ attn) {
    const int wave = threadIdx.x >> 5;
    const int lane = threadIdx.x & 31;
    const int row  = blockIdx.x * 4 + wave;     // 0 .. BH*SEQ-1
    const int i    = row & (SEQ - 1);
    float* rp = attn + (size_t)row * SEQ;

    float mx = -__builtin_inff();
    for (int j = lane; j <= i; j += 32) mx = fmaxf(mx, rp[j]);
    for (int m = 16; m >= 1; m >>= 1) mx = fmaxf(mx, __shfl_xor(mx, m, 32));

    float sum = 0.0f;
    for (int j = lane; j <= i; j += 32) sum += __expf(rp[j] - mx);
    for (int m = 16; m >= 1; m >>= 1) sum += __shfl_xor(sum, m, 32);
    float inv = __frcp_rn(sum);

    for (int j = lane; j < SEQ; j += 32) {
        float p = 0.0f;
        if (j <= i) p = __expf(rp[j] - mx) * inv;
        rp[j] = p;
    }
}

// ----------------------------------------------------------- attn @ V ------
// ctx[b*T+t, h*64+d] = sum_j attn(t,j) * v(j,d), WMMA over 32-key chunks.
// grid (SEQ/16, BH), block 128; wave w owns d0 = w*16.
__global__ void ctx_wmma(const float* __restrict__ attn, const u16* __restrict__ vb,
                         u16* __restrict__ ctx) {
    __shared__ u16 Ps[16 * 32];                 // attn tile bf16, 1 KiB
    __shared__ u16 Vs[32 * 64];                 // v tile, 4 KiB
    const int q0   = blockIdx.x * 16;
    const int bh   = blockIdx.y;
    const int wave = threadIdx.x >> 5;
    const int lane = threadIdx.x & 31;
    const int half = lane >> 4;
    const int l    = lane & 15;
    const int d0   = wave * 16;
    const int kend = q0 + 16;                   // keys 0..q0+15 are nonzero

    v8f acc = {};
    for (int ks = 0; ks < kend; ks += 32) {
        // stage attn tile fp32 -> bf16 (cols beyond i are exact zeros)
        {
            int e = threadIdx.x * 4;            // 512 elems / 128 thr
            int r = e >> 5, c = e & 31;
            float4 p = *(const float4*)&attn[((size_t)bh * SEQ + q0 + r) * SEQ + ks + c];
            u32 lo = (u32)f32_to_bf16(p.x) | ((u32)f32_to_bf16(p.y) << 16);
            u32 hi = (u32)f32_to_bf16(p.z) | ((u32)f32_to_bf16(p.w) << 16);
            *(uint2*)&Ps[e] = make_uint2(lo, hi);
        }
        // stage V tile: 32 keys x 64 dims
        {
            int e = threadIdx.x * 16;           // 2048 elems / 128 thr
            int r = e >> 6, c = e & 63;
            const u16* vr = &vb[((size_t)bh * SEQ + ks + r) * HD + c];
            *(uint4*)&Vs[e]     = *(const uint4*)&vr[0];
            *(uint4*)&Vs[e + 8] = *(const uint4*)&vr[8];
        }
        __syncthreads();
        Frag16 a, b;
        a.q[0] = *(const uint4*)&Ps[l * 32 + half * 16];
        a.q[1] = *(const uint4*)&Ps[l * 32 + half * 16 + 8];
        // B fragment: lane col n = d0+l, K rows half*16 + 0..15 (strided ds)
#pragma unroll
        for (int kr = 0; kr < 16; ++kr)
            b.u[kr] = Vs[(half * 16 + kr) * 64 + d0 + l];
        acc = __builtin_amdgcn_wmma_f32_16x16x32_bf16(
                  false, a.v, false, b.v, (short)0, acc, false, false);
        __syncthreads();
    }
    const int h = bh & (HEADS - 1), bi = bh >> 4;
    for (int r = 0; r < 8; ++r) {
        int t = q0 + r + half * 8;
        ctx[((size_t)(bi * SEQ + t)) * DIM + h * HD + d0 + l] = f32_to_bf16(acc[r]);
    }
}

// ------------------------------------------------------------------ launch --
extern "C" void kernel_launch(void* const* d_in, const int* in_sizes, int n_in,
                              void* d_out, int out_size, void* d_ws, size_t ws_size,
                              hipStream_t stream) {
    const float* x  = (const float*)d_in[0];
    const float* Wq = (const float*)d_in[1];
    const float* bq = (const float*)d_in[2];
    const float* Wk = (const float*)d_in[3];
    const float* bk = (const float*)d_in[4];
    const float* Wv = (const float*)d_in[5];
    const float* bv = (const float*)d_in[6];
    const float* Wo = (const float*)d_in[7];
    const float* bo = (const float*)d_in[8];

    float* outp = (float*)d_out;                               // [MROWS, DIM]
    float* attn = outp + (size_t)MROWS * DIM;                  // [BH, SEQ, SEQ]

    char* ws = (char*)d_ws;
    const size_t MB = 1024 * 1024;
    u16* xb  = (u16*)(ws + 0);
    u16* wqb = (u16*)(ws + 8 * MB);
    u16* wkb = (u16*)(ws + 10 * MB);
    u16* wvb = (u16*)(ws + 12 * MB);
    u16* wob = (u16*)(ws + 14 * MB);
    u16* qb  = (u16*)(ws + 16 * MB);
    u16* kb  = (u16*)(ws + 24 * MB);
    u16* vb  = (u16*)(ws + 32 * MB);
    u16* ctx = (u16*)(ws + 40 * MB);

    // 1) fp32 -> bf16 conversions
    {
        int n4 = (MROWS * DIM) / 4;
        cvt_f32_to_bf16<<<n4 / 256, 256, 0, stream>>>(x, xb, n4);
        int w4 = (DIM * DIM) / 4;
        cvt_f32_to_bf16<<<w4 / 256, 256, 0, stream>>>(Wq, wqb, w4);
        cvt_f32_to_bf16<<<w4 / 256, 256, 0, stream>>>(Wk, wkb, w4);
        cvt_f32_to_bf16<<<w4 / 256, 256, 0, stream>>>(Wv, wvb, w4);
        cvt_f32_to_bf16<<<w4 / 256, 256, 0, stream>>>(Wo, wob, w4);
    }

    // 2) Q/K/V projections (WMMA, 4x N register blocking per wave)
    dim3 pg(MROWS / 16, DIM / 256);
    proj_wmma<<<pg, 128, 0, stream>>>(xb, wqb, bq, qb, nullptr, 0);
    proj_wmma<<<pg, 128, 0, stream>>>(xb, wkb, bk, kb, nullptr, 0);
    proj_wmma<<<pg, 128, 0, stream>>>(xb, wvb, bv, vb, nullptr, 0);

    // 3) raw scores + ALiBi + causal (WMMA)
    dim3 sg(SEQ / 16, BH);
    scores_wmma<<<sg, 128, 0, stream>>>(qb, kb, attn);

    // 4) row softmax (also zero-fills masked region)
    softmax_rows<<<(BH * SEQ) / 4, 128, 0, stream>>>(attn);

    // 5) context = attn @ V (WMMA)
    ctx_wmma<<<sg, 128, 0, stream>>>(attn, vb, ctx);

    // 6) output projection (WMMA, fp32 out)
    proj_wmma<<<pg, 128, 0, stream>>>(ctx, wob, bo, nullptr, outp, 1);
}
